// get_model_pred_84610855731459
// MI455X (gfx1250) — matrix operand order repaired and checked by reading, not compiled, not run on hardware
//
#include <hip/hip_runtime.h>

typedef __bf16 bf16_t;
typedef __attribute__((ext_vector_type(16))) __bf16 v16bf;
typedef __attribute__((ext_vector_type(8)))  __bf16 v8bf;
typedef __attribute__((ext_vector_type(8)))  float  v8f;

#define ALPC 2.2f
#define BELC 0.025f

// ---------------- prep: W[k][n] (f32) -> WT[n][k] (bf16) ----------------
__global__ void transpose_to_bf16(const float* __restrict__ W, bf16_t* __restrict__ WT,
                                  int K, int N) {
    int idx = blockIdx.x * blockDim.x + threadIdx.x;
    if (idx >= K * N) return;
    int k = idx / N, n = idx % N;
    WT[n * K + k] = (bf16_t)W[k * N + n];
}

// ---------------- fragment loaders (bf16 16x16x32 layouts) ----------------
// A: 16x32, lane holds row m=lane&15; K chunks {half*8..half*8+7} and {16+half*8..}
__device__ inline v16bf load_frag_a(const bf16_t* base, int stride, int row0, int k0, int lane) {
    int m = lane & 15, half = lane >> 4;
    const bf16_t* p = base + (row0 + m) * stride + k0 + half * 8;
    union { v16bf v; v8bf h[2]; } u;
    u.h[0] = *(const v8bf*)(p);
    u.h[1] = *(const v8bf*)(p + 16);
    return u.v;
}
// B: 32x16 from transposed weight WT[n][k]; lane holds col n=lane&15,
// K chunks {half*8..half*8+7} and {16+half*8..}
__device__ inline v16bf load_frag_b(const bf16_t* wt, int strideK, int n0, int k0, int lane) {
    int n = lane & 15, half = lane >> 4;
    const bf16_t* p = wt + (size_t)(n0 + n) * strideK + k0 + half * 8;
    union { v16bf v; v8bf h[2]; } u;
    u.h[0] = *(const v8bf*)(p);
    u.h[1] = *(const v8bf*)(p + 16);
    return u.v;
}

__device__ inline float gate_fn(float p) {
    const float hi = 1.0f / ALPC + BELC;
    return (p <= BELC) ? 0.0f : ((p >= hi) ? 1.0f : ALPC * (p - BELC));
}

// ---------------- fused MLP + type head + softmax + gate ----------------
__global__ __launch_bounds__(256) void fused_mlp(
    const float* __restrict__ pred,        // [E][512]
    const float* __restrict__ be1, const float* __restrict__ be2,
    const float* __restrict__ bt1, const float* __restrict__ bt2,
    const float* __restrict__ Wt2,         // [128][4] f32
    const bf16_t* __restrict__ We1T,       // [512][512]
    const bf16_t* __restrict__ We2T,       // [512][512]
    const bf16_t* __restrict__ Wt1T,       // [128][512]
    float* __restrict__ out, int E)
{
    constexpr int D = 512, S = 520, MT = 128;
    extern __shared__ char smem[];
    bf16_t* Abf  = (bf16_t*)smem;                                   // X tile, then E tile
    bf16_t* Hbf  = (bf16_t*)(smem + (size_t)MT * S * sizeof(bf16_t)); // H tile
    float*  Wt2l = (float*)(smem + (size_t)2 * MT * S * sizeof(bf16_t)); // 512 f32
    float*  T1f  = (float*)Hbf;            // reuse: [128][132] f32
    float*  Plds = (float*)Hbf + 128 * 132; // [128][4]

    const int tid  = threadIdx.x;
    const int wave = tid >> 5;
    const int lane = tid & 31;
    const int row0 = blockIdx.x * MT;
    const size_t Es = (size_t)E;
    float* out_tw = out;                    // types_w  [4][E]
    float* out_mw = out + 4 * Es;           // multiW   [4][E]
    float* out_to = out + 8 * Es;           // type_out [E][4]
    float* out_ee = out + 12 * Es;          // edge_emb [E][512]

    // stage Wt2 into LDS
    for (int i = tid; i < 512; i += 256) Wt2l[i] = Wt2[i];

    // cooperative load of X tile -> bf16 LDS (clamp rows for last block)
    for (int idx = tid; idx < MT * (D / 4); idx += 256) {
        int r  = idx / (D / 4);
        int c4 = (idx % (D / 4)) * 4;
        int gr = row0 + r; if (gr >= E) gr = E - 1;
        float4 x = *(const float4*)(pred + (size_t)gr * D + c4);
        union { bf16_t b[4]; uint2 u; } pk;
        pk.b[0] = (bf16_t)x.x; pk.b[1] = (bf16_t)x.y;
        pk.b[2] = (bf16_t)x.z; pk.b[3] = (bf16_t)x.w;
        *(uint2*)(Abf + r * S + c4) = pk.u;
    }
    __syncthreads();

    const int mt = wave;                 // 16-row stripe per wave
    const int nl = lane & 15, half = lane >> 4;

    // ---- GEMM1: H = relu(X @ We1 + be1) ; 32 col-tiles in groups of 4 ----
    for (int g = 0; g < 8; ++g) {
        v8f acc[4] = {};
        for (int k0 = 0; k0 < D; k0 += 32) {
            v16bf a = load_frag_a(Abf, S, mt * 16, k0, lane);
#pragma unroll
            for (int t = 0; t < 4; ++t) {
                v16bf b = load_frag_b(We1T, D, (g * 4 + t) * 16, k0, lane);
                acc[t] = __builtin_amdgcn_wmma_f32_16x16x32_bf16(
                    false, a, false, b, (short)0, acc[t], false, false);
            }
        }
#pragma unroll
        for (int t = 0; t < 4; ++t) {
            int n0 = (g * 4 + t) * 16;
            float bv = be1[n0 + nl];
#pragma unroll
            for (int r = 0; r < 8; ++r) {
                int m = half * 8 + r;
                float v = acc[t][r] + bv;
                v = v > 0.0f ? v : 0.0f;
                Hbf[(mt * 16 + m) * S + n0 + nl] = (bf16_t)v;
            }
        }
    }
    __syncthreads();

    // ---- GEMM2: Ee = H @ We2 + be2 ; write f32 out + bf16 back into Abf ----
    for (int g = 0; g < 8; ++g) {
        v8f acc[4] = {};
        for (int k0 = 0; k0 < D; k0 += 32) {
            v16bf a = load_frag_a(Hbf, S, mt * 16, k0, lane);
#pragma unroll
            for (int t = 0; t < 4; ++t) {
                v16bf b = load_frag_b(We2T, D, (g * 4 + t) * 16, k0, lane);
                acc[t] = __builtin_amdgcn_wmma_f32_16x16x32_bf16(
                    false, a, false, b, (short)0, acc[t], false, false);
            }
        }
#pragma unroll
        for (int t = 0; t < 4; ++t) {
            int n0 = (g * 4 + t) * 16;
            float bv = be2[n0 + nl];
#pragma unroll
            for (int r = 0; r < 8; ++r) {
                int m = half * 8 + r;
                float v = acc[t][r] + bv;
                int gm = row0 + mt * 16 + m;
                if (gm < E) out_ee[(size_t)gm * D + n0 + nl] = v;
                Abf[(mt * 16 + m) * S + n0 + nl] = (bf16_t)v;
            }
        }
    }
    __syncthreads();

    // ---- GEMM3: T1 = relu(Ee @ Wt1 + bt1) ; 8 col-tiles in 2 groups ----
    for (int g = 0; g < 2; ++g) {
        v8f acc[4] = {};
        for (int k0 = 0; k0 < D; k0 += 32) {
            v16bf a = load_frag_a(Abf, S, mt * 16, k0, lane);
#pragma unroll
            for (int t = 0; t < 4; ++t) {
                v16bf b = load_frag_b(Wt1T, D, (g * 4 + t) * 16, k0, lane);
                acc[t] = __builtin_amdgcn_wmma_f32_16x16x32_bf16(
                    false, a, false, b, (short)0, acc[t], false, false);
            }
        }
#pragma unroll
        for (int t = 0; t < 4; ++t) {
            int n0 = (g * 4 + t) * 16;
            float bv = bt1[n0 + nl];
#pragma unroll
            for (int r = 0; r < 8; ++r) {
                int m = half * 8 + r;
                float v = acc[t][r] + bv;
                v = v > 0.0f ? v : 0.0f;
                T1f[(mt * 16 + m) * 132 + n0 + nl] = v;
            }
        }
    }
    __syncthreads();

    // ---- final 128->4 layer: each thread does 2 of the 4 outputs of one row ----
    {
        int m  = tid & 127;
        int jp = tid >> 7;                 // 0 -> j{0,1}, 1 -> j{2,3}
        float a0 = bt2[2 * jp], a1 = bt2[2 * jp + 1];
        const float* trow = T1f + m * 132;
        for (int k = 0; k < 128; ++k) {
            float t = trow[k];
            a0 += t * Wt2l[k * 4 + 2 * jp];
            a1 += t * Wt2l[k * 4 + 2 * jp + 1];
        }
        Plds[m * 4 + 2 * jp]     = a0;
        Plds[m * 4 + 2 * jp + 1] = a1;
    }
    __syncthreads();

    // ---- softmax + types_w + gate, one row per thread ----
    if (tid < 128) {
        int gm = row0 + tid;
        if (gm < E) {
            float p0 = Plds[tid * 4 + 0], p1 = Plds[tid * 4 + 1];
            float p2 = Plds[tid * 4 + 2], p3 = Plds[tid * 4 + 3];
            float mx = fmaxf(fmaxf(p0, p1), fmaxf(p2, p3));
            float e0 = __expf(p0 - mx), e1 = __expf(p1 - mx);
            float e2 = __expf(p2 - mx), e3 = __expf(p3 - mx);
            float inv = 1.0f / (e0 + e1 + e2 + e3);
            float t0 = e0 * inv, t1 = e1 * inv, t2 = e2 * inv, t3 = e3 * inv;
            float4 to = make_float4(t0, t1, t2, t3);
            *(float4*)(out_to + (size_t)gm * 4) = to;
            float tw0 = 1.0f - t0;
            out_tw[0 * Es + gm] = tw0; out_tw[1 * Es + gm] = t1;
            out_tw[2 * Es + gm] = t2;  out_tw[3 * Es + gm] = t3;
            out_mw[0 * Es + gm] = gate_fn(tw0); out_mw[1 * Es + gm] = gate_fn(t1);
            out_mw[2 * Es + gm] = gate_fn(t2);  out_mw[3 * Es + gm] = gate_fn(t3);
        }
    }
}

extern "C" void kernel_launch(void* const* d_in, const int* in_sizes, int n_in,
                              void* d_out, int out_size, void* d_ws, size_t ws_size,
                              hipStream_t stream) {
    // inputs: 0 obj_codes (unused), 1 pred_codes, 2 We1, 3 be1, 4 We2, 5 be2,
    //         6 Wt1, 7 bt1, 8 Wt2, 9 bt2
    const float* pred = (const float*)d_in[1];
    const float* We1  = (const float*)d_in[2];
    const float* be1  = (const float*)d_in[3];
    const float* We2  = (const float*)d_in[4];
    const float* be2  = (const float*)d_in[5];
    const float* Wt1  = (const float*)d_in[6];
    const float* bt1  = (const float*)d_in[7];
    const float* Wt2  = (const float*)d_in[8];
    const float* bt2  = (const float*)d_in[9];
    const int D = 512, H = 128;
    const int E = in_sizes[1] / D;

    bf16_t* We1T = (bf16_t*)d_ws;
    bf16_t* We2T = We1T + (size_t)D * D;
    bf16_t* Wt1T = We2T + (size_t)D * D;

    transpose_to_bf16<<<(D * D + 255) / 256, 256, 0, stream>>>(We1, We1T, D, D);
    transpose_to_bf16<<<(D * D + 255) / 256, 256, 0, stream>>>(We2, We2T, D, D);
    transpose_to_bf16<<<(D * H + 255) / 256, 256, 0, stream>>>(Wt1, Wt1T, D, H);

    const int MT = 128, S = 520;
    size_t lds_bytes = (size_t)2 * MT * S * sizeof(bf16_t) + 512 * sizeof(float);
    int nblocks = (E + MT - 1) / MT;
    fused_mlp<<<nblocks, 256, lds_bytes, stream>>>(
        pred, be1, be2, bt1, bt2, Wt2, We1T, We2T, Wt1T, (float*)d_out, E);
}
